// MHA_63763084476883
// MI455X (gfx1250) — matrix operand order repaired and checked by reading, not compiled, not run on hardware
//
#include <hip/hip_runtime.h>

// Problem constants (match reference)
#define B_   2
#define S_   2048
#define D_   1024
#define QKV_ 2048
#define H_   16
#define HD_  128
#define SCALE_ 0.08838834764831845f  // 128^-0.5

typedef __attribute__((ext_vector_type(16))) __bf16          v16bf;
typedef __attribute__((ext_vector_type(8)))  float           v8f;
typedef __attribute__((ext_vector_type(8)))  unsigned short  ushort8v;
typedef __attribute__((ext_vector_type(4)))  unsigned int    uint4v;
typedef __attribute__((ext_vector_type(4)))  int             int4v;
typedef __attribute__((ext_vector_type(8)))  int             int8v;

#if defined(__gfx1250__) && __has_builtin(__builtin_amdgcn_tensor_load_to_lds) && \
    __has_builtin(__builtin_amdgcn_s_wait_tensorcnt)
#define HAVE_TDM 1
#else
#define HAVE_TDM 0
#endif

struct Frag2 { ushort8v lo, hi; };

__device__ __forceinline__ unsigned short f2bf(float f) {
  union { float f; unsigned u; } c; c.f = f;
  return (unsigned short)((c.u + 0x7FFFu + ((c.u >> 16) & 1u)) >> 16);
}

__device__ __forceinline__ v8f wmma_bf16(v16bf a, v16bf b, v8f c) {
  return __builtin_amdgcn_wmma_f32_16x16x32_bf16(false, a, false, b, (short)0, c, false, false);
}

// A fragment (16x32 bf16): lane holds row m = lane&15; hi = lane>>4.
__device__ __forceinline__ v16bf load_frag_a(const unsigned short* p, int ld, int kBase, int lane) {
  int m = lane & 15, hi = lane >> 4;
  const unsigned short* row = p + m * ld + kBase + hi * 8;
  Frag2 f;
  f.lo = *(const ushort8v*)(row);
  f.hi = *(const ushort8v*)(row + 16);
  return __builtin_bit_cast(v16bf, f);
}

// B fragment (32x16 bf16): source holds B^T row-major (src[n][k], ld elems).
__device__ __forceinline__ v16bf load_frag_b(const unsigned short* p, int ld, int kBase, int lane) {
  int n = lane & 15, hi = lane >> 4;
  const unsigned short* row = p + n * ld + kBase + hi * 8;
  Frag2 f;
  f.lo = *(const ushort8v*)(row);
  f.hi = *(const ushort8v*)(row + 16);
  return __builtin_bit_cast(v16bf, f);
}

#if HAVE_TDM
// TDM: DMA a [tile_h x tile_w] tile of 2-byte elements (row stride = stride
// elements in global memory) into LDS at byte offset lds_off, packed row-major.
// D# layout per CDNA5 ISA ch.8 (group0: flags/lds/global/type, group1: dims).
// Toolchain is the 6-arg builtin: (v4u g0, v8i g1, v4i g2, v4i g3, v8i, i32 cpol).
__device__ __forceinline__ void tdm_load_2d_bf16(unsigned lds_off, const void* gsrc,
                                                 int tile_w, int tile_h, long long stride) {
  unsigned long long ga = (unsigned long long)gsrc;
  uint4v g0;
  g0[0] = 1u;                                               // count=1, user D#
  g0[1] = lds_off;                                          // LDS byte address
  g0[2] = (unsigned)(ga & 0xFFFFFFFFull);                   // global_addr[31:0]
  g0[3] = (unsigned)((ga >> 32) & 0x1FFFFFFull) | (2u << 30); // [56:32] | type=2
  int8v g1;
  g1[0] = (int)(1u << 16);                                  // data_size = 2 bytes
  g1[1] = (int)((unsigned)tile_w << 16);                    // tensor_dim0[15:0]
  g1[2] = (int)((unsigned)tile_h << 16);                    // tensor_dim1[15:0]
  g1[3] = (int)((unsigned)tile_w << 16);                    // tile_dim0
  g1[4] = (int)(unsigned)tile_h;                            // tile_dim1 (dim2=0)
  g1[5] = (int)(unsigned)(stride & 0xFFFFFFFFll);           // dim0_stride[31:0]
  g1[6] = (int)(unsigned)((stride >> 32) & 0xFFFFll);       // dim0_stride[47:32]
  g1[7] = 0;
  int4v gz4 = {0, 0, 0, 0};
  int8v gz8 = (int8v)(0);
  __builtin_amdgcn_tensor_load_to_lds(g0, g1, gz4, gz4, gz8, 0);
}
#endif

// ---------------------------------------------------------------------------
// Tiled GEMM: C[M,N] = A[M,K] * W[N,K]^T, 128x128 tile, K-step 32, 8 waves.
// Register-pipelined: tile t+1 global loads overlap tile t WMMAs.
// CMODE: 0 = f32 row-major, 1 = bf16 row-major, 2 = bf16 per-head transposed
//        (V stored as [B,H,HD,S] so attention P*V B-frags are contiguous).
// ---------------------------------------------------------------------------
#define TM 128
#define TN 128
#define TK 32

template <bool A_IS_F32, int CMODE>
__global__ __launch_bounds__(256) void gemm_wmma(const void* __restrict__ Aptr,
                                                 const float* __restrict__ Bw,
                                                 void* __restrict__ Cptr,
                                                 int M, int N, int K, float scale) {
  __shared__ unsigned short sA[TM * TK];  // 8 KB
  __shared__ unsigned short sB[TN * TK];  // 8 KB

  const int tid   = threadIdx.x;
  const int lane  = tid & 31;
  const int wave  = tid >> 5;
  const int waveM = wave >> 1;   // 0..3
  const int waveN = wave & 1;    // 0..1
  const int mBase = blockIdx.y * TM;
  const int nBase = blockIdx.x * TN;

  const int row = tid >> 1;            // 0..127
  const int seg = (tid & 1) * 16;      // 0 or 16
  const size_t aOff = (size_t)(mBase + row) * K + seg;
  const size_t bOff = (size_t)(nBase + row) * K + seg;

  v8f acc[2][4];
#pragma unroll
  for (int i = 0; i < 2; ++i)
#pragma unroll
    for (int j = 0; j < 4; ++j) acc[i][j] = (v8f)(0.0f);

  float aF[16], bF[16];
  ushort8v aH0, aH1;
  (void)aH0; (void)aH1;

  // ---- preload k0 = 0 into registers ----
  if constexpr (A_IS_F32) {
    const float* s = (const float*)Aptr + aOff;
#pragma unroll
    for (int u = 0; u < 16; ++u) aF[u] = s[u];
  }
#if !HAVE_TDM
  else {
    const unsigned short* s = (const unsigned short*)Aptr + aOff;
    aH0 = *(const ushort8v*)s; aH1 = *(const ushort8v*)(s + 8);
  }
#endif
  {
    const float* s = Bw + bOff;
#pragma unroll
    for (int u = 0; u < 16; ++u) bF[u] = s[u];
  }

  for (int k0 = 0; k0 < K; k0 += TK) {
    // ---- commit staged registers to LDS (or DMA A tile via TDM) ----
    if constexpr (A_IS_F32) {
      ushort8v t0, t1;
#pragma unroll
      for (int u = 0; u < 8; ++u) { t0[u] = f2bf(aF[u]); t1[u] = f2bf(aF[u + 8]); }
      *(ushort8v*)&sA[row * TK + seg]     = t0;
      *(ushort8v*)&sA[row * TK + seg + 8] = t1;
    } else {
#if HAVE_TDM
      if (wave == 0)
        tdm_load_2d_bf16((unsigned)(size_t)&sA[0],
                         (const unsigned short*)Aptr + (size_t)mBase * K + k0,
                         TK, TM, (long long)K);
#else
      *(ushort8v*)&sA[row * TK + seg]     = aH0;
      *(ushort8v*)&sA[row * TK + seg + 8] = aH1;
#endif
    }
    {
      ushort8v t0, t1;
#pragma unroll
      for (int u = 0; u < 8; ++u) { t0[u] = f2bf(bF[u]); t1[u] = f2bf(bF[u + 8]); }
      *(ushort8v*)&sB[row * TK + seg]     = t0;
      *(ushort8v*)&sB[row * TK + seg + 8] = t1;
    }
#if HAVE_TDM
    if constexpr (!A_IS_F32) {
      if (wave == 0) __builtin_amdgcn_s_wait_tensorcnt(0);
    }
#endif
    __syncthreads();

    // ---- issue next tile's global loads (overlaps the WMMAs below) ----
    if (k0 + TK < K) {
      if constexpr (A_IS_F32) {
        const float* s = (const float*)Aptr + aOff + k0 + TK;
#pragma unroll
        for (int u = 0; u < 16; ++u) aF[u] = s[u];
      }
#if !HAVE_TDM
      else {
        const unsigned short* s = (const unsigned short*)Aptr + aOff + k0 + TK;
        aH0 = *(const ushort8v*)s; aH1 = *(const ushort8v*)(s + 8);
      }
#endif
      const float* s = Bw + bOff + k0 + TK;
#pragma unroll
      for (int u = 0; u < 16; ++u) bF[u] = s[u];
    }

    // ---- WMMA burst ----
    v16bf af[2], bfb[4];
#pragma unroll
    for (int i = 0; i < 2; ++i)
      af[i] = load_frag_a(&sA[(waveM * 32 + i * 16) * TK], TK, 0, lane);
#pragma unroll
    for (int j = 0; j < 4; ++j)
      bfb[j] = load_frag_b(&sB[(waveN * 64 + j * 16) * TK], TK, 0, lane);
#pragma unroll
    for (int i = 0; i < 2; ++i)
#pragma unroll
      for (int j = 0; j < 4; ++j)
        acc[i][j] = wmma_bf16(af[i], bfb[j], acc[i][j]);
    __syncthreads();
  }

  // ---- epilogue ----
  const int n = lane & 15, hi = lane >> 4;
#pragma unroll
  for (int i = 0; i < 2; ++i)
#pragma unroll
    for (int j = 0; j < 4; ++j)
#pragma unroll
      for (int r = 0; r < 8; ++r) {
        int gm = mBase + waveM * 32 + i * 16 + r + 8 * hi;
        int gn = nBase + waveN * 64 + j * 16 + n;
        float v = acc[i][j][r] * scale;
        if constexpr (CMODE == 0) {
          ((float*)Cptr)[(size_t)gm * N + gn] = v;
        } else if constexpr (CMODE == 1) {
          ((unsigned short*)Cptr)[(size_t)gm * N + gn] = f2bf(v);
        } else {
          // V^T layout: [B, H, HD, S]
          int bb = gm >> 11, srow = gm & (S_ - 1);
          int hh = gn >> 7,  d    = gn & (HD_ - 1);
          ((unsigned short*)Cptr)[(((size_t)bb * H_ + hh) * HD_ + d) * S_ + srow] = f2bf(v);
        }
      }
}

// ---------------------------------------------------------------------------
// Flash attention (causal, online softmax). Grid: (S/128, H, B), 8 waves.
// K tiles (64x128 bf16) DMA'd into LDS by the Tensor Data Mover; V read as
// B-frags directly from the [B,H,HD,S] transposed global layout (L2-resident).
// ---------------------------------------------------------------------------
#define KT 64

__global__ __launch_bounds__(256) void flash_attn(const unsigned short* __restrict__ Q,
                                                  const unsigned short* __restrict__ Kb,
                                                  const unsigned short* __restrict__ VT,
                                                  unsigned short* __restrict__ Att) {
  __shared__ unsigned short sK[KT * HD_];     // [key][d]   16 KB
  __shared__ unsigned short sP[8 * 16 * KT];  // per-wave P 16 KB

  const int tid  = threadIdx.x;
  const int lane = tid & 31;
  const int wave = tid >> 5;
  const int n    = lane & 15;
  const int hi   = lane >> 4;

  const int qTile = blockIdx.x;
  const int h     = blockIdx.y;
  const int b     = blockIdx.z;
  const int qBase = qTile * 128;
  const int qRow0 = qBase + wave * 16;

  const int krow = tid >> 2;          // 0..63   (staging / prefetch mapping)
  const int c0   = (tid & 3) * 32;    // 0,32,64,96

  const unsigned short* qp = Q + ((size_t)(b * S_ + qRow0)) * QKV_ + h * HD_;
  const unsigned short* vt = VT + ((size_t)(b * H_ + h)) * HD_ * S_;  // [HD][S]

  v16bf qf[4];
#pragma unroll
  for (int kk = 0; kk < 4; ++kk) qf[kk] = load_frag_a(qp, QKV_, kk * 32, lane);

  v8f o[8];
#pragma unroll
  for (int df = 0; df < 8; ++df) o[df] = (v8f)(0.0f);
  float mr[8], lr[8];
#pragma unroll
  for (int r = 0; r < 8; ++r) { mr[r] = -1e30f; lr[r] = 0.0f; }

  const int ktiles = (qBase + 128) / KT;
  for (int t = 0; t < ktiles; ++t) {
    const int kBase = t * KT;
    __syncthreads();  // previous tile's LDS reads complete

    // ---- stage K tile [64 x 128] into LDS ----
#if HAVE_TDM
    if (wave == 0) {
      tdm_load_2d_bf16((unsigned)(size_t)&sK[0],
                       Kb + ((size_t)(b * S_ + kBase)) * QKV_ + h * HD_,
                       HD_, KT, (long long)QKV_);
      __builtin_amdgcn_s_wait_tensorcnt(0);
    }
#else
    {
      const unsigned short* ks = Kb + ((size_t)(b * S_ + kBase + krow)) * QKV_ + h * HD_ + c0;
#pragma unroll
      for (int u = 0; u < 4; ++u)
        *(ushort8v*)&sK[krow * HD_ + c0 + u * 8] = *(const ushort8v*)(ks + u * 8);
    }
#endif
    __syncthreads();

    // prefetch next K tile into L2 while computing this one
    if (t + 1 < ktiles)
      __builtin_prefetch(Kb + ((size_t)(b * S_ + kBase + KT + krow)) * QKV_ + h * HD_ + c0, 0, 1);

    // ---- S = Q K^T (16 queries x 64 keys per wave) ----
    v8f s[4];
#pragma unroll
    for (int nf = 0; nf < 4; ++nf) s[nf] = (v8f)(0.0f);
#pragma unroll
    for (int nf = 0; nf < 4; ++nf)
#pragma unroll
      for (int kk = 0; kk < 4; ++kk)
        s[nf] = wmma_bf16(qf[kk], load_frag_b(&sK[nf * 16 * HD_], HD_, kk * 32, lane), s[nf]);

    // ---- causal mask + online softmax ----
    float nm[8];
#pragma unroll
    for (int r = 0; r < 8; ++r) nm[r] = mr[r];
#pragma unroll
    for (int nf = 0; nf < 4; ++nf) {
      const int key = kBase + nf * 16 + n;
#pragma unroll
      for (int r = 0; r < 8; ++r) {
        const int q = qRow0 + r + 8 * hi;
        float v = (key <= q) ? s[nf][r] : -1e30f;
        s[nf][r] = v;
        nm[r] = fmaxf(nm[r], v);
      }
    }
#pragma unroll
    for (int msk = 1; msk < 16; msk <<= 1)
#pragma unroll
      for (int r = 0; r < 8; ++r) nm[r] = fmaxf(nm[r], __shfl_xor(nm[r], msk, 16));

    float al[8], rs[8];
#pragma unroll
    for (int r = 0; r < 8; ++r) { al[r] = __expf(mr[r] - nm[r]); mr[r] = nm[r]; rs[r] = 0.0f; }
#pragma unroll
    for (int nf = 0; nf < 4; ++nf)
#pragma unroll
      for (int r = 0; r < 8; ++r) {
        float p = __expf(s[nf][r] - nm[r]);
        s[nf][r] = p;
        rs[r] += p;
      }
#pragma unroll
    for (int msk = 1; msk < 16; msk <<= 1)
#pragma unroll
      for (int r = 0; r < 8; ++r) rs[r] += __shfl_xor(rs[r], msk, 16);
#pragma unroll
    for (int r = 0; r < 8; ++r) lr[r] = lr[r] * al[r] + rs[r];
#pragma unroll
    for (int df = 0; df < 8; ++df)
#pragma unroll
      for (int r = 0; r < 8; ++r) o[df][r] *= al[r];

    // ---- P -> per-wave LDS (bf16), reload as A-frags (same-wave DS in-order) ----
    unsigned short* pw = &sP[wave * 16 * KT];
#pragma unroll
    for (int nf = 0; nf < 4; ++nf)
#pragma unroll
      for (int r = 0; r < 8; ++r)
        pw[(r + 8 * hi) * KT + nf * 16 + n] = f2bf(s[nf][r]);

    // ---- O += P * V : B-frags straight from global V^T [HD][S] ----
#pragma unroll
    for (int kk = 0; kk < 2; ++kk) {
      v16bf pa = load_frag_a(pw, KT, kk * 32, lane);
#pragma unroll
      for (int df = 0; df < 8; ++df)
        o[df] = wmma_bf16(pa, load_frag_b(vt + (size_t)(df * 16) * S_, S_, kBase + kk * 32, lane), o[df]);
    }
  }

  // ---- epilogue: O / l -> att[b, q, h*HD + d] (bf16) ----
#pragma unroll
  for (int df = 0; df < 8; ++df)
#pragma unroll
    for (int r = 0; r < 8; ++r) {
      const int qrow = qRow0 + r + 8 * hi;
      const int col  = h * HD_ + df * 16 + n;
      Att[((size_t)(b * S_ + qrow)) * QKV_ + col] = f2bf(o[df][r] / lr[r]);
    }
}

// ---------------------------------------------------------------------------
extern "C" void kernel_launch(void* const* d_in, const int* in_sizes, int n_in,
                              void* d_out, int out_size, void* d_ws, size_t ws_size,
                              hipStream_t stream) {
  (void)in_sizes; (void)n_in; (void)out_size; (void)ws_size;
  const float* x  = (const float*)d_in[0];
  const float* wq = (const float*)d_in[1];
  const float* wk = (const float*)d_in[2];
  const float* wv = (const float*)d_in[3];
  const float* wo = (const float*)d_in[4];
  float* out = (float*)d_out;

  const size_t NTOK = (size_t)B_ * S_;   // 4096
  const size_t QKVE = NTOK * QKV_;       // 8,388,608 elems per buffer

  unsigned short* qb   = (unsigned short*)d_ws;
  unsigned short* kb   = qb + QKVE;
  unsigned short* vtb  = kb + QKVE;      // V stored transposed: [B,H,HD,S]
  unsigned short* attb = vtb + QKVE;     // total 64 MB of ws

  dim3 blk(256);

  // QKV projections: [4096,1024] x [2048,1024]^T
  dim3 gQKV(QKV_ / TN, (int)(NTOK / TM));  // (16, 32)
  gemm_wmma<true, 1><<<gQKV, blk, 0, stream>>>(x, wq, qb,  (int)NTOK, QKV_, D_, SCALE_);
  gemm_wmma<true, 1><<<gQKV, blk, 0, stream>>>(x, wk, kb,  (int)NTOK, QKV_, D_, 1.0f);
  gemm_wmma<true, 2><<<gQKV, blk, 0, stream>>>(x, wv, vtb, (int)NTOK, QKV_, D_, 1.0f);

  // Causal flash attention
  dim3 gATT(S_ / 128, H_, B_);             // (16, 16, 2)
  flash_attn<<<gATT, blk, 0, stream>>>(qb, kb, vtb, attb);

  // Output projection: [4096,2048] x [1024,2048]^T -> f32
  dim3 gOUT(D_ / TN, (int)(NTOK / TM));    // (8, 32)
  gemm_wmma<false, 0><<<gOUT, blk, 0, stream>>>(attb, wo, out, (int)NTOK, D_, QKV_, 1.0f);
}
